// Attention_5514738008849
// MI455X (gfx1250) — matrix operand order repaired and checked by reading, not compiled, not run on hardware
//
#include <hip/hip_runtime.h>

typedef __attribute__((ext_vector_type(16))) __bf16 v16bf;
typedef __attribute__((ext_vector_type(8)))  float  v8f;
typedef __attribute__((__vector_size__(16))) int    v4i;

union FragU { v16bf v; uint4 q[2]; };
union QuadU { uint4 q; unsigned short h[8]; };

// ---- native bf16 conversion ----
__device__ __forceinline__ unsigned short f2bf(float x){
  __bf16 b = (__bf16)x;
  return __builtin_bit_cast(unsigned short, b);
}
__device__ __forceinline__ unsigned int f2bf2(float lo, float hi){
  return (unsigned int)f2bf(lo) | ((unsigned int)f2bf(hi) << 16);
}
__device__ __forceinline__ float bf2f(unsigned short h){
  return __uint_as_float(((unsigned int)h) << 16);
}

__device__ __forceinline__ v8f wmma_bf16(v16bf a, v16bf b, v8f c){
  return __builtin_amdgcn_wmma_f32_16x16x32_bf16(false, a, false, b, (short)0, c, false, false);
}

// A-operand 16x32 bf16: lane lid = M row; halves: K in [hi*8,+8) then [16+hi*8,+8)
__device__ __forceinline__ v16bf load_fragA(const unsigned short* p, int stride, int lane){
  int lid = lane & 15, hi = (lane >> 4) & 1;
  const unsigned short* r = p + lid * stride + hi * 8;
  FragU f;
  f.q[0] = *(const uint4*)(r);
  f.q[1] = *(const uint4*)(r + 16);
  return f.v;
}
// B-operand 32x16 bf16 stored [n][k]: lane lid = N col; halves: K in [hi*8,+8) then [16+hi*8,+8)
__device__ __forceinline__ v16bf load_fragB(const unsigned short* p, int stride, int lane){
  int lid = lane & 15, hi = (lane >> 4) & 1;
  const unsigned short* r = p + lid * stride + hi * 8;
  FragU f;
  f.q[0] = *(const uint4*)(r);
  f.q[1] = *(const uint4*)(r + 16);
  return f.v;
}

// butterfly reductions within each 16-lane half
#define SWZF(x, m) __int_as_float(__builtin_amdgcn_ds_swizzle(__float_as_int(x), (0x1f | ((m) << 10))))
__device__ __forceinline__ float redmax16(float x){
  x = fmaxf(x, SWZF(x, 1)); x = fmaxf(x, SWZF(x, 2));
  x = fmaxf(x, SWZF(x, 4)); x = fmaxf(x, SWZF(x, 8));
  return x;
}
__device__ __forceinline__ float redsum16(float x){
  x += SWZF(x, 1); x += SWZF(x, 2); x += SWZF(x, 4); x += SWZF(x, 8);
  return x;
}

// ---- CDNA5 async global->LDS copy (ASYNCcnt path) ----
#if defined(__has_builtin)
# if __has_builtin(__builtin_amdgcn_global_load_async_to_lds_b128)
#  define USE_ASYNC 1
# endif
#endif
#ifndef USE_ASYNC
# define USE_ASYNC 0
#endif

#if USE_ASYNC
#define AS1 __attribute__((address_space(1)))
#define AS3 __attribute__((address_space(3)))
__device__ __forceinline__ void async_copy16(const void* g, void* l){
  __builtin_amdgcn_global_load_async_to_lds_b128(
      (AS1 v4i*)(unsigned long long)(size_t)g,
      (AS3 v4i*)(unsigned int)(size_t)l, 0, 0);
}
__device__ __forceinline__ void async_wait0(){
# if __has_builtin(__builtin_amdgcn_s_wait_asynccnt)
  __builtin_amdgcn_s_wait_asynccnt(0);
# else
  asm volatile("s_wait_asynccnt 0" ::: "memory");
# endif
}
#endif

// ---------------- Kernel 1: QKV GEMM, double-buffered LDS ----------------
__global__ __launch_bounds__(256) void qkv_gemm_kernel(const float* __restrict__ x,
                                                       const float* __restrict__ w,
                                                       unsigned short* __restrict__ qkv){
  __shared__ unsigned short As[2][128][32];
  __shared__ unsigned short Bs[2][128][32];
  const int tid = threadIdx.x;
  const int lane = tid & 31, wave = tid >> 5;
  const int lid = lane & 15, hi = (lane >> 4) & 1;
  const int bm = blockIdx.x * 128, bn = blockIdx.y * 128;
  const int wm = (wave & 1) * 64, wn = (wave >> 1) * 32;
  v8f acc[4][2] = {};
  float4 ar[4], br[4];

#pragma unroll
  for (int j = 0; j < 4; ++j){
    int idx = tid + 256 * j;
    ar[j] = *(const float4*)(x + (size_t)(bm + (idx >> 3)) * 1024 + (idx & 7) * 4);
    br[j] = *(const float4*)(w + (size_t)(idx >> 5) * 3072 + bn + (idx & 31) * 4);
  }
#pragma unroll
  for (int j = 0; j < 4; ++j){
    int idx = tid + 256 * j;
    *(uint2*)&As[0][idx >> 3][(idx & 7) * 4] = make_uint2(f2bf2(ar[j].x, ar[j].y), f2bf2(ar[j].z, ar[j].w));
    int n4 = (idx & 31) * 4, k = idx >> 5;
    Bs[0][n4 + 0][k] = f2bf(br[j].x); Bs[0][n4 + 1][k] = f2bf(br[j].y);
    Bs[0][n4 + 2][k] = f2bf(br[j].z); Bs[0][n4 + 3][k] = f2bf(br[j].w);
  }
  __syncthreads();

  for (int t = 0; t < 32; ++t){
    const int cur = t & 1;
    const int kc_next = (t + 1) * 32;
    if (t + 1 < 32){
#pragma unroll
      for (int j = 0; j < 4; ++j){
        int idx = tid + 256 * j;
        ar[j] = *(const float4*)(x + (size_t)(bm + (idx >> 3)) * 1024 + kc_next + (idx & 7) * 4);
        br[j] = *(const float4*)(w + (size_t)(kc_next + (idx >> 5)) * 3072 + bn + (idx & 31) * 4);
      }
    }
    v16bf a[4], b[2];
#pragma unroll
    for (int i = 0; i < 4; ++i) a[i] = load_fragA(&As[cur][wm + 16 * i][0], 32, lane);
#pragma unroll
    for (int j = 0; j < 2; ++j) b[j] = load_fragB(&Bs[cur][wn + 16 * j][0], 32, lane);
#pragma unroll
    for (int i = 0; i < 4; ++i)
#pragma unroll
      for (int j = 0; j < 2; ++j)
        acc[i][j] = wmma_bf16(a[i], b[j], acc[i][j]);
    if (t + 1 < 32){
      const int nxt = cur ^ 1;
#pragma unroll
      for (int j = 0; j < 4; ++j){
        int idx = tid + 256 * j;
        *(uint2*)&As[nxt][idx >> 3][(idx & 7) * 4] = make_uint2(f2bf2(ar[j].x, ar[j].y), f2bf2(ar[j].z, ar[j].w));
        int n4 = (idx & 31) * 4, k = idx >> 5;
        Bs[nxt][n4 + 0][k] = f2bf(br[j].x); Bs[nxt][n4 + 1][k] = f2bf(br[j].y);
        Bs[nxt][n4 + 2][k] = f2bf(br[j].z); Bs[nxt][n4 + 3][k] = f2bf(br[j].w);
      }
    }
    __syncthreads();
  }
#pragma unroll
  for (int i = 0; i < 4; ++i)
#pragma unroll
    for (int j = 0; j < 2; ++j)
#pragma unroll
      for (int e = 0; e < 8; ++e){
        int row = bm + wm + 16 * i + e + 8 * hi;
        int col = bn + wn + 16 * j + lid;
        int s = col >> 10, rem = col & 1023;
        int hh = rem >> 6, dd = rem & 63;
        int bb = row >> 10, nn = row & 1023;
        size_t addr = ((((size_t)s * 8 + bb) * 16 + hh) * 1024 + nn) * 64 + dd;
        qkv[addr] = f2bf(acc[i][j][e]);
      }
}

// ---------------- Kernel 2: axial RoPE in-place on Q and K ----------------
__global__ __launch_bounds__(256) void rope_kernel(unsigned short* __restrict__ qkv){
  const int idx = blockIdx.x * 256 + threadIdx.x;
  unsigned int* prow = (unsigned int*)(qkv + (size_t)idx * 64);
  const int n = idx & 1023;
  const int hp = n >> 5, wp = n & 31;
  const float posH = -1.0f + (2.0f / 31.0f) * (float)hp;
  const float posW = -1.0f + (2.0f / 31.0f) * (float)wp;
#pragma unroll
  for (int i = 0; i < 16; ++i){
    float base = (1.0f + (511.0f / 7.0f) * (float)(i & 7)) * 3.14159265358979f;
    float f = ((i < 8) ? posH : posW) * base;
    float c = __cosf(f), sn = __sinf(f);
    unsigned int pv = prow[i];
    float x1 = bf2f((unsigned short)(pv & 0xffffu));
    float x2 = bf2f((unsigned short)(pv >> 16));
    prow[i] = f2bf2(x1 * c - x2 * sn, x2 * c + x1 * sn);
  }
}

// ---------------- Kernel 3: flash attention per (b, h, 64-row q-tile) ----------------
__global__ __launch_bounds__(128) void attn_kernel(const unsigned short* __restrict__ qkv,
                                                   unsigned short* __restrict__ attn){
  __shared__ unsigned short Ks[32][64];      // [key][d]
  __shared__ unsigned short Vs[64][32];      // [d][key]  (transposed)
  __shared__ unsigned short Ps[4][16][32];   // per-wave P staging
  const int tid = threadIdx.x;
  const int lane = tid & 31, wave = tid >> 5;
  const int lid = lane & 15, hi = (lane >> 4) & 1;
  const int blk = blockIdx.x;
  const int qt = blk & 15, h = (blk >> 4) & 15, b = blk >> 8;
  const size_t headQ = (((size_t)(0 * 8 + b) * 16 + h) * 1024) * 64;
  const size_t headK = (((size_t)(1 * 8 + b) * 16 + h) * 1024) * 64;
  const size_t headV = (((size_t)(2 * 8 + b) * 16 + h) * 1024) * 64;
  const unsigned short* Qp = qkv + headQ + (size_t)(qt * 64 + wave * 16) * 64;
  v16bf qf0 = load_fragA(Qp, 64, lane);
  v16bf qf1 = load_fragA(Qp + 32, 64, lane);
  v8f o[4] = {};
  float m[8], l[8];
#pragma unroll
  for (int e = 0; e < 8; ++e){ m[e] = -1e30f; l[e] = 0.0f; }
  for (int kc = 0; kc < 1024; kc += 32){
#pragma unroll
    for (int j = 0; j < 2; ++j){               // K tile: async copy if available
      int idx = tid + 128 * j;
      int key = idx >> 3, d8 = (idx & 7) * 8;
#if USE_ASYNC
      async_copy16(qkv + headK + (size_t)(kc + key) * 64 + d8, &Ks[key][d8]);
#else
      *(uint4*)&Ks[key][d8] = *(const uint4*)(qkv + headK + (size_t)(kc + key) * 64 + d8);
#endif
      QuadU v; v.q = *(const uint4*)(qkv + headV + (size_t)(kc + key) * 64 + d8);
#pragma unroll
      for (int t = 0; t < 8; ++t) Vs[d8 + t][key] = v.h[t];
    }
#if USE_ASYNC
    async_wait0();
#endif
    __syncthreads();
    if (kc + 32 < 1024) __builtin_prefetch(qkv + headK + (size_t)(kc + 32) * 64, 0, 1);
    v8f s0 = {}, s1 = {};
    {
      v16bf k00 = load_fragB(&Ks[0][0], 64, lane);
      v16bf k01 = load_fragB(&Ks[0][32], 64, lane);
      s0 = wmma_bf16(qf0, k00, s0);
      s0 = wmma_bf16(qf1, k01, s0);
      v16bf k10 = load_fragB(&Ks[16][0], 64, lane);
      v16bf k11 = load_fragB(&Ks[16][32], 64, lane);
      s1 = wmma_bf16(qf0, k10, s1);
      s1 = wmma_bf16(qf1, k11, s1);
    }
    const float scale = 0.125f;
#pragma unroll
    for (int e = 0; e < 8; ++e){
      float a = s0[e] * scale, c = s1[e] * scale;
      float mx = redmax16(fmaxf(a, c));
      float mn = fmaxf(m[e], mx);
      float alpha = __expf(m[e] - mn);
      float pa = __expf(a - mn), pb = __expf(c - mn);
      float rs = redsum16(pa + pb);
      l[e] = l[e] * alpha + rs;
      m[e] = mn;
#pragma unroll
      for (int t = 0; t < 4; ++t) o[t][e] *= alpha;
      Ps[wave][e + 8 * hi][lid] = f2bf(pa);
      Ps[wave][e + 8 * hi][16 + lid] = f2bf(pb);
    }
    asm volatile("s_wait_dscnt 0" ::: "memory");
    v16bf pf = load_fragA(&Ps[wave][0][0], 32, lane);
#pragma unroll
    for (int t = 0; t < 4; ++t){
      v16bf vf = load_fragB(&Vs[t * 16][0], 32, lane);
      o[t] = wmma_bf16(pf, vf, o[t]);
    }
    __syncthreads();
  }
#pragma unroll
  for (int t = 0; t < 4; ++t)
#pragma unroll
    for (int e = 0; e < 8; ++e){
      float val = o[t][e] / l[e];
      int n = qt * 64 + wave * 16 + e + 8 * hi;
      int col = h * 64 + t * 16 + lid;
      attn[((size_t)b * 1024 + n) * 1024 + col] = f2bf(val);
    }
}

// ---------------- Kernel 4: projection GEMM + bias, double-buffered ----------------
__global__ __launch_bounds__(256) void proj_kernel(const unsigned short* __restrict__ a,
                                                   const float* __restrict__ w,
                                                   const float* __restrict__ bias,
                                                   float* __restrict__ out){
  __shared__ unsigned short As[2][128][32];
  __shared__ unsigned short Bs[2][128][32];
  const int tid = threadIdx.x;
  const int lane = tid & 31, wave = tid >> 5;
  const int lid = lane & 15, hi = (lane >> 4) & 1;
  const int bm = blockIdx.x * 128, bn = blockIdx.y * 128;
  const int wm = (wave & 1) * 64, wn = (wave >> 1) * 32;
  v8f acc[4][2] = {};
  uint4 ar[2]; float4 br[4];

#pragma unroll
  for (int j = 0; j < 2; ++j){
    int idx = tid + 256 * j;
    ar[j] = *(const uint4*)(a + (size_t)(bm + (idx >> 2)) * 1024 + (idx & 3) * 8);
  }
#pragma unroll
  for (int j = 0; j < 4; ++j){
    int idx = tid + 256 * j;
    br[j] = *(const float4*)(w + (size_t)(idx >> 5) * 1024 + bn + (idx & 31) * 4);
  }
#pragma unroll
  for (int j = 0; j < 2; ++j){
    int idx = tid + 256 * j;
    *(uint4*)&As[0][idx >> 2][(idx & 3) * 8] = ar[j];
  }
#pragma unroll
  for (int j = 0; j < 4; ++j){
    int idx = tid + 256 * j;
    int n4 = (idx & 31) * 4, k = idx >> 5;
    Bs[0][n4 + 0][k] = f2bf(br[j].x); Bs[0][n4 + 1][k] = f2bf(br[j].y);
    Bs[0][n4 + 2][k] = f2bf(br[j].z); Bs[0][n4 + 3][k] = f2bf(br[j].w);
  }
  __syncthreads();

  for (int t = 0; t < 32; ++t){
    const int cur = t & 1;
    const int kc_next = (t + 1) * 32;
    if (t + 1 < 32){
#pragma unroll
      for (int j = 0; j < 2; ++j){
        int idx = tid + 256 * j;
        ar[j] = *(const uint4*)(a + (size_t)(bm + (idx >> 2)) * 1024 + kc_next + (idx & 3) * 8);
      }
#pragma unroll
      for (int j = 0; j < 4; ++j){
        int idx = tid + 256 * j;
        br[j] = *(const float4*)(w + (size_t)(kc_next + (idx >> 5)) * 1024 + bn + (idx & 31) * 4);
      }
    }
    v16bf af[4], bf[2];
#pragma unroll
    for (int i = 0; i < 4; ++i) af[i] = load_fragA(&As[cur][wm + 16 * i][0], 32, lane);
#pragma unroll
    for (int j = 0; j < 2; ++j) bf[j] = load_fragB(&Bs[cur][wn + 16 * j][0], 32, lane);
#pragma unroll
    for (int i = 0; i < 4; ++i)
#pragma unroll
      for (int j = 0; j < 2; ++j)
        acc[i][j] = wmma_bf16(af[i], bf[j], acc[i][j]);
    if (t + 1 < 32){
      const int nxt = cur ^ 1;
#pragma unroll
      for (int j = 0; j < 2; ++j){
        int idx = tid + 256 * j;
        *(uint4*)&As[nxt][idx >> 2][(idx & 3) * 8] = ar[j];
      }
#pragma unroll
      for (int j = 0; j < 4; ++j){
        int idx = tid + 256 * j;
        int n4 = (idx & 31) * 4, k = idx >> 5;
        Bs[nxt][n4 + 0][k] = f2bf(br[j].x); Bs[nxt][n4 + 1][k] = f2bf(br[j].y);
        Bs[nxt][n4 + 2][k] = f2bf(br[j].z); Bs[nxt][n4 + 3][k] = f2bf(br[j].w);
      }
    }
    __syncthreads();
  }
  float bv[2];
#pragma unroll
  for (int j = 0; j < 2; ++j) bv[j] = bias[bn + wn + 16 * j + lid];
#pragma unroll
  for (int i = 0; i < 4; ++i)
#pragma unroll
    for (int j = 0; j < 2; ++j)
#pragma unroll
      for (int e = 0; e < 8; ++e){
        int row = bm + wm + 16 * i + e + 8 * hi;
        int col = bn + wn + 16 * j + lid;
        out[(size_t)row * 1024 + col] = acc[i][j][e] + bv[j];
      }
}

extern "C" void kernel_launch(void* const* d_in, const int* in_sizes, int n_in,
                              void* d_out, int out_size, void* d_ws, size_t ws_size,
                              hipStream_t stream){
  (void)in_sizes; (void)n_in; (void)out_size; (void)ws_size;
  const float* x      = (const float*)d_in[0];
  const float* w_qkv  = (const float*)d_in[1];
  const float* w_proj = (const float*)d_in[2];
  const float* b_proj = (const float*)d_in[3];
  float* out = (float*)d_out;
  unsigned short* qkv  = (unsigned short*)d_ws;
  unsigned short* attn = (unsigned short*)((char*)d_ws + (size_t)50331648);

  qkv_gemm_kernel<<<dim3(64, 24), 256, 0, stream>>>(x, w_qkv, qkv);
  rope_kernel<<<dim3(1024), 256, 0, stream>>>(qkv);
  attn_kernel<<<dim3(2048), 128, 0, stream>>>(qkv, attn);
  proj_kernel<<<dim3(64, 8), 256, 0, stream>>>(attn, w_proj, b_proj, out);
}
